// GALENetwork3D_19301583028427
// MI455X (gfx1250) — matrix operand order repaired and checked by reading, not compiled, not run on hardware
//
#include <hip/hip_runtime.h>
#include <hip/hip_bf16.h>
#include <stdint.h>

typedef __attribute__((ext_vector_type(16))) _Float16 v16h;
typedef __attribute__((ext_vector_type(2)))  _Float16 v2h;
typedef __attribute__((ext_vector_type(2)))  __fp16   v2fp16;
typedef __attribute__((ext_vector_type(8)))  float    v8f;
typedef __attribute__((ext_vector_type(4)))  float    v4f;
typedef __attribute__((ext_vector_type(2)))  float    v2f;
typedef __attribute__((ext_vector_type(4)))  int      v4i;

#define BLOCK       256
#define NW          (BLOCK / 32)
#define TW          2                       // row-tiles (of 16) per wave
#define ROWS_PER_WG (NW * 16 * TW)          // 256
#define HDIM        64
#define PERT        0.02f
#define ALPHA       0.6f

// Per-layer weight record (f16 weights + f32 biases), used both in the
// workspace (pre-converted) and as the LDS staging format:
//   halves [0..255]     W1p [64][4]   k=0..2 -> W1 row n, k=3 -> b1[n] (bias folded)
//   halves [256..4351]  W2  [64][64]  row n = W2[n][*]  (B-operand form)
//   halves [4352..4543] W3  [3][64]
//   byte 9088: float[68]: [0..63]=b2, [64..66]=b3, [67]=pad
#define REC_W1          0
#define REC_W2          256
#define REC_W3          4352
#define REC_HALVES      4544
#define REC_BIAS_BYTES  (REC_HALVES * 2)        // 9088
#define REC_BYTES       (REC_BIAS_BYTES + 68*4) // 9360 (16B multiple)

#if defined(__gfx1250__) && __has_builtin(__builtin_amdgcn_global_load_async_to_lds_b128)
#define STAGE_ASYNC_HW 1
#else
#define STAGE_ASYNC_HW 0
#endif

__device__ __forceinline__ void wait_async0() {
#if __has_builtin(__builtin_amdgcn_s_wait_asynccnt)
  __builtin_amdgcn_s_wait_asynccnt(0);
#else
  asm volatile("s_wait_asynccnt 0x0" ::: "memory");
#endif
}

// relu in a single VALU op: v_med3_f32(x, 0, +big) == clamp(x, 0, big)
__device__ __forceinline__ float relu1(float x) {
  return __builtin_amdgcn_fmed3f(x, 0.0f, 3.0e38f);
}

// pack two f32 into two f16 halves with one v_cvt_pk_rtz_f16_f32
__device__ __forceinline__ v2h pack2(float a, float b) {
#if __has_builtin(__builtin_amdgcn_cvt_pkrtz)
  union { v2fp16 s; v2h d; } u;
  u.s = __builtin_amdgcn_cvt_pkrtz(a, b);
  return u.d;
#else
  v2h r; r.x = (_Float16)a; r.y = (_Float16)b; return r;
#endif
}

// Copy one 9360B weight record global -> LDS. Async (ASYNCcnt-tracked) when the
// gfx1250 builtin is available; plain 16B vector copy otherwise.
__device__ __forceinline__ void stage_copy(const char* __restrict__ gsrc,
                                           char* ldst, int tid) {
#if STAGE_ASYNC_HW
  for (int off = tid * 16; off < REC_BYTES; off += BLOCK * 16) {
    __builtin_amdgcn_global_load_async_to_lds_b128(
        (v4i*)(gsrc + off), (v4i*)(ldst + off), 0, 0);
  }
#else
  for (int off = tid * 16; off < REC_BYTES; off += BLOCK * 16)
    *reinterpret_cast<v4f*>(ldst + off) =
        *reinterpret_cast<const v4f*>(gsrc + off);
#endif
}

// ---- CDNA5 WMMA fragment loaders (wave32, v_wmma_f32_16x16x32_f16) --------
// A 16x32: lanes 0-15 row M=lane, K=k0+0..15;
//          lanes 16-31 row M=lane-16, K=k0+16..31.
__device__ __forceinline__ v16h lds_load_Afrag(const _Float16* base /*[16][64]*/,
                                               int k0, int lane) {
  const int mm = lane & 15;
  const int ko = k0 + (lane & 16);
  union { v16h h; v4f f[2]; } u;
  u.f[0] = *reinterpret_cast<const v4f*>(base + mm * HDIM + ko);
  u.f[1] = *reinterpret_cast<const v4f*>(base + mm * HDIM + ko + 8);
  return u.h;
}

// B 32x16: lanes 0-15 col N=n0+lane, K=k0+0..15; lanes 16-31 col N=n0+lane-16,
// K=k0+16..31. B=W^T, so col n of B = row n of W (contiguous row-major reads).
__device__ __forceinline__ v16h lds_load_BfragW2(const _Float16* W, int n0,
                                                 int k0, int lane) {
  const int nn = n0 + (lane & 15);
  const int kk = k0 + (lane & 16);
  const _Float16* p = W + nn * HDIM + kk;
  union { v16h h; v4f f[2]; } u;
  u.f[0] = *reinterpret_cast<const v4f*>(p);
  u.f[1] = *reinterpret_cast<const v4f*>(p + 8);
  return u.h;
}

// W1p is [64][4] (k=0..2 weights, k=3 bias); K=4..31 are implicit zeros.
__device__ __forceinline__ v16h lds_load_BfragW1(const _Float16* W1p, int n0,
                                                 int lane) {
  const int nn = n0 + (lane & 15);
  v2f w = *reinterpret_cast<const v2f*>(W1p + nn * 4);
  v2f z = {0.f, 0.f};
  union { v16h h; v2f g[4]; } u;
  u.g[0] = (lane & 16) ? z : w;   // hi half-wave covers K=8..15,24..31 -> zero
  u.g[1] = z; u.g[2] = z; u.g[3] = z;
  return u.h;
}

// W3 is stored as its 3 real rows [3][64]; columns N=3..15 are zeros.
__device__ __forceinline__ v16h lds_load_BfragW3(const _Float16* W3, int k0,
                                                 int lane) {
  const int nn = lane & 15;
  const int kk = k0 + (lane & 16);
  const _Float16* p = W3 + (nn < 3 ? nn : 0) * HDIM + kk;
  v4f f0 = *reinterpret_cast<const v4f*>(p);
  v4f f1 = *reinterpret_cast<const v4f*>(p + 8);
  v4f z = {0.f, 0.f, 0.f, 0.f};
  const bool okn = nn < 3;
  union { v16h h; v4f f[2]; } u;
  u.f[0] = okn ? f0 : z;
  u.f[1] = okn ? f1 : z;
  return u.h;
}

__device__ __forceinline__ v8f wmma(v16h a, v16h b, v8f c) {
  return __builtin_amdgcn_wmma_f32_16x16x32_f16(false, a, false, b,
                                                (short)0, c, false, false);
}

template <bool ASYNC>
__global__ __launch_bounds__(BLOCK)
void gale_net3d_kernel(const float* __restrict__ g_x,
                       const float* __restrict__ g_W1, const float* __restrict__ g_b1,
                       const float* __restrict__ g_W2, const float* __restrict__ g_b2,
                       const float* __restrict__ g_W3, const float* __restrict__ g_b3,
                       const float* __restrict__ g_noise,
                       const char* __restrict__ g_wrec,   // pre-converted records (ASYNC)
                       float* __restrict__ g_out,
                       int n, int L)
{
  __shared__ alignas(16) char     s_rec[2][REC_BYTES];        // double-buffered weights
  __shared__ alignas(16) _Float16 s_h[NW][TW][16 * HDIM];     // activation shuffle scratch
  __shared__ float                s_d[NW][TW][16][4];         // delta scatter

  const int tid    = threadIdx.x;
  const int lane   = tid & 31;
  const int wv     = tid >> 5;
  const int m      = lane & 15;
  const int hiHalf = (lane >> 4) & 1;
  const int mB     = hiHalf * 8;

  const long long rowBase = (long long)blockIdx.x * ROWS_PER_WG + wv * (16 * TW);
  long long g[TW], gc[TW];
  bool ok[TW];
  float xc[TW][3], cx[TW][3];
#pragma unroll
  for (int j = 0; j < TW; ++j) {
    g[j]  = rowBase + j * 16 + m;
    ok[j] = g[j] < (long long)n;
    gc[j] = ok[j] ? g[j] : 0;
#pragma unroll
    for (int c = 0; c < 3; ++c) {
      xc[j][c] = g_x[gc[j] * 3 + c];
      cx[j][c] = xc[j][c];
    }
  }

  if (ASYNC) stage_copy(g_wrec, s_rec[0], tid);   // prologue: layer 0

  for (int l = 0; l < L; ++l) {
    int b;
    if (ASYNC) {
      wait_async0();          // this wave's pending copies for s_rec[l&1] done
      __syncthreads();        // all waves' copies done; prev layer compute done
      if (l + 1 < L)          // overlap next layer's staging with this compute
        stage_copy(g_wrec + (size_t)(l + 1) * REC_BYTES, s_rec[(l + 1) & 1], tid);
      b = l & 1;
    } else {
      __syncthreads();        // protect s_rec[0] from readers of previous layer
      _Float16* w = (_Float16*)s_rec[0];
      for (int i = tid; i < 256; i += BLOCK) {
        int nn = i >> 2, k = i & 3;
        w[REC_W1 + i] =
            (_Float16)((k < 3) ? g_W1[(l * 64 + nn) * 3 + k] : g_b1[l * 64 + nn]);
      }
      for (int i = tid; i < 4096; i += BLOCK)
        w[REC_W2 + i] = (_Float16)g_W2[(size_t)l * 4096 + i];
      for (int i = tid; i < 192; i += BLOCK)
        w[REC_W3 + i] = (_Float16)g_W3[l * 192 + i];
      float* bb = (float*)(s_rec[0] + REC_BIAS_BYTES);
      if (tid < 64)       bb[tid] = g_b2[l * 64 + tid];
      else if (tid < 67)  bb[tid] = g_b3[l * 3 + (tid - 64)];
      else if (tid == 67) bb[tid] = 0.f;
      __syncthreads();
      b = 0;
    }

    const _Float16* W1p = (const _Float16*)s_rec[b] + REC_W1;
    const _Float16* W2p = (const _Float16*)s_rec[b] + REC_W2;
    const _Float16* W3p = (const _Float16*)s_rec[b] + REC_W3;
    const float*    bsp = (const float*)(s_rec[b] + REC_BIAS_BYTES);

    // per-row noise for this layer; prefetch next layer's rows
    float nz[TW][3];
#pragma unroll
    for (int j = 0; j < TW; ++j) {
      const size_t nb = ((size_t)l * (size_t)n + (size_t)gc[j]) * 3;
      nz[j][0] = g_noise[nb + 0];
      nz[j][1] = g_noise[nb + 1];
      nz[j][2] = g_noise[nb + 2];
      if (l + 1 < L)
        __builtin_prefetch(
            g_noise + ((size_t)(l + 1) * (size_t)n + (size_t)gc[j]) * 3, 0, 0);
    }

    // A fragments of x: K=0..2 coords, K=3 = 1.0 (bias lane), hi half zero
    v16h ax[TW];
#pragma unroll
    for (int j = 0; j < TW; ++j) {
      v16h a = {};
      a[0] = (_Float16)(hiHalf ? 0.f : xc[j][0]);
      a[1] = (_Float16)(hiHalf ? 0.f : xc[j][1]);
      a[2] = (_Float16)(hiHalf ? 0.f : xc[j][2]);
      a[3] = (_Float16)(hiHalf ? 0.f : 1.0f);
      ax[j] = a;
    }

    _Float16* h0 = s_h[wv][0];
    _Float16* h1 = s_h[wv][1];

    // ---- h1 = relu(x @ W1^T + b1)  (bias folded into K=3) ----
#pragma unroll
    for (int t = 0; t < 4; ++t) {
      v16h bf = lds_load_BfragW1(W1p, 16 * t, lane);
      v8f z = {0.f,0.f,0.f,0.f,0.f,0.f,0.f,0.f};
      v8f acc0 = wmma(ax[0], bf, z);
      v8f acc1 = wmma(ax[1], bf, z);
      const int nc = 16 * t + m;
#pragma unroll
      for (int r = 0; r < 8; ++r) {
        v2h pk = pack2(relu1(acc0[r]), relu1(acc1[r]));
        h0[(mB + r) * HDIM + nc] = pk.x;
        h1[(mB + r) * HDIM + nc] = pk.y;
      }
    }

    // C-layout -> A-layout via wave-private LDS (DS in-order within a wave)
    v16h a00 = lds_load_Afrag(h0, 0, lane);
    v16h a01 = lds_load_Afrag(h0, 32, lane);
    v16h a10 = lds_load_Afrag(h1, 0, lane);
    v16h a11 = lds_load_Afrag(h1, 32, lane);

    // ---- h2 = relu(h1 @ W2^T + b2): bias rides in as the WMMA C operand ----
#pragma unroll
    for (int t = 0; t < 4; ++t) {
      v16h b0 = lds_load_BfragW2(W2p, 16 * t, 0, lane);
      v16h b1 = lds_load_BfragW2(W2p, 16 * t, 32, lane);
      const float bias = bsp[16 * t + m];
      v8f cb = {bias, bias, bias, bias, bias, bias, bias, bias};
      v8f acc0 = wmma(a00, b0, cb);
      v8f acc1 = wmma(a10, b0, cb);
      acc0 = wmma(a01, b1, acc0);
      acc1 = wmma(a11, b1, acc1);
      const int nc = 16 * t + m;
#pragma unroll
      for (int r = 0; r < 8; ++r) {
        v2h pk = pack2(relu1(acc0[r]), relu1(acc1[r]));
        h0[(mB + r) * HDIM + nc] = pk.x;
        h1[(mB + r) * HDIM + nc] = pk.y;
      }
    }

    v16h c00 = lds_load_Afrag(h0, 0, lane);
    v16h c01 = lds_load_Afrag(h0, 32, lane);
    v16h c10 = lds_load_Afrag(h1, 0, lane);
    v16h c11 = lds_load_Afrag(h1, 32, lane);

    // ---- delta = h2 @ W3^T : one 16-col tile (cols 0..2 valid), K=64 ----
    v16h w30 = lds_load_BfragW3(W3p, 0, lane);
    v16h w31 = lds_load_BfragW3(W3p, 32, lane);
    v8f zz = {0.f,0.f,0.f,0.f,0.f,0.f,0.f,0.f};
    v8f d0 = wmma(c00, w30, zz);
    v8f d1 = wmma(c10, w30, zz);
    d0 = wmma(c01, w31, d0);
    d1 = wmma(c11, w31, d1);

    if (m < 3) {
#pragma unroll
      for (int r = 0; r < 8; ++r) {
        s_d[wv][0][mB + r][m] = d0[r];
        s_d[wv][1][mB + r][m] = d1[r];
      }
    }

    const float b30 = bsp[64], b31 = bsp[65], b32v = bsp[66];
#pragma unroll
    for (int j = 0; j < TW; ++j) {
      const float dd0 = s_d[wv][j][m][0];
      const float dd1 = s_d[wv][j][m][1];
      const float dd2 = s_d[wv][j][m][2];
      const float p0 = xc[j][0] + PERT * (dd0 + b30) + nz[j][0];
      const float p1 = xc[j][1] + PERT * (dd1 + b31) + nz[j][1];
      const float p2 = xc[j][2] + PERT * (dd2 + b32v) + nz[j][2];
      xc[j][0] = (1.0f - ALPHA) * p0 + ALPHA * cx[j][0];
      xc[j][1] = (1.0f - ALPHA) * p1 + ALPHA * cx[j][1];
      xc[j][2] = (1.0f - ALPHA) * p2 + ALPHA * cx[j][2];
    }
  }

  if (lane < 16) {
#pragma unroll
    for (int j = 0; j < TW; ++j)
      if (ok[j]) {
        g_out[g[j] * 3 + 0] = xc[j][0];
        g_out[g[j] * 3 + 1] = xc[j][1];
        g_out[g[j] * 3 + 2] = xc[j][2];
      }
  }
}

// One-shot f32 -> f16 weight pre-conversion into the workspace record format.
__global__ void convert_weights_kernel(const float* __restrict__ W1,
                                       const float* __restrict__ b1,
                                       const float* __restrict__ W2,
                                       const float* __restrict__ b2,
                                       const float* __restrict__ W3,
                                       const float* __restrict__ b3,
                                       char* __restrict__ ws, int L) {
  const int per = REC_HALVES + 68;
  const int idx = blockIdx.x * blockDim.x + threadIdx.x;
  if (idx >= L * per) return;
  const int l = idx / per;
  const int e = idx % per;
  char* rec = ws + (size_t)l * REC_BYTES;
  if (e < REC_HALVES) {
    float v;
    if (e < 256) {
      int nn = e >> 2, k = e & 3;
      v = (k < 3) ? W1[(l * 64 + nn) * 3 + k] : b1[l * 64 + nn];
    } else if (e < 4352) {
      v = W2[(size_t)l * 4096 + (e - 256)];
    } else {
      v = W3[l * 192 + (e - 4352)];
    }
    ((_Float16*)rec)[e] = (_Float16)v;
  } else {
    const int i = e - REC_HALVES;
    const float v = (i < 64) ? b2[l * 64 + i]
                             : (i < 67 ? b3[l * 3 + (i - 64)] : 0.f);
    ((float*)(rec + REC_BIAS_BYTES))[i] = v;
  }
}

extern "C" void kernel_launch(void* const* d_in, const int* in_sizes, int n_in,
                              void* d_out, int out_size, void* d_ws, size_t ws_size,
                              hipStream_t stream) {
  const float* x  = (const float*)d_in[0];
  const float* W1 = (const float*)d_in[1];
  const float* b1 = (const float*)d_in[2];
  const float* W2 = (const float*)d_in[3];
  const float* b2 = (const float*)d_in[4];
  const float* W3 = (const float*)d_in[5];
  const float* b3 = (const float*)d_in[6];
  const float* nz = (const float*)d_in[7];
  float* out = (float*)d_out;

  const int n = in_sizes[0] / 3;      // number of points
  const int L = in_sizes[2] / 64;     // b1 is [L, 64]
  const int grid = (n + ROWS_PER_WG - 1) / ROWS_PER_WG;
  const size_t wsNeed = (size_t)L * REC_BYTES;

  if (ws_size >= wsNeed) {
    const int tot = L * (REC_HALVES + 68);
    hipLaunchKernelGGL(convert_weights_kernel, dim3((tot + 255) / 256), dim3(256),
                       0, stream, W1, b1, W2, b2, W3, b3, (char*)d_ws, L);
    hipLaunchKernelGGL((gale_net3d_kernel<true>), dim3(grid), dim3(BLOCK), 0,
                       stream, x, W1, b1, W2, b2, W3, b3, nz,
                       (const char*)d_ws, out, n, L);
  } else {
    hipLaunchKernelGGL((gale_net3d_kernel<false>), dim3(grid), dim3(BLOCK), 0,
                       stream, x, W1, b1, W2, b2, W3, b3, nz,
                       (const char*)nullptr, out, n, L);
  }
}
